// BetterGooLayer_44092134261064
// MI455X (gfx1250) — matrix operand order repaired and checked by reading, not compiled, not run on hardware
//
#include <hip/hip_runtime.h>
#include <stdint.h>

typedef __attribute__((ext_vector_type(16))) _Float16 v16h;
typedef __attribute__((ext_vector_type(8)))  float    v8f;

#define NS      8192
#define CHUNK   128
#define NCHUNK  64
#define NBM     256      /* B*M */
#define DAMP    0.9998f

/* workspace layout (floats) */
#define WS_X      0
#define WS_ZEND   (2097152)
#define WS_SSTART (2097152 + 262144)

/* d_out layout (floats): recording | displacement | hf */
#define OUT_REC  0
#define OUT_DISP (2097152)
#define OUT_HF   (2097152 + 16777216)

// ---------------------------------------------------------------------------
// Pass 1: each (chain, chunk) runs 128 steps from zero state; store end state.
// ---------------------------------------------------------------------------
__global__ void k_chunk_zero(const float* __restrict__ forces,
                             const float* __restrict__ hm,
                             const float* __restrict__ home,
                             const float* __restrict__ cs,
                             const float* __restrict__ masses,
                             const float* __restrict__ tensions,
                             float* __restrict__ zend)
{
    int idx   = blockIdx.x * blockDim.x + threadIdx.x;   // 131072 tasks
    int chunk = idx & (NCHUNK - 1);
    int chain = idx >> 6;
    int d = chain & 7;
    int m = (chain >> 3) & 63;
    float a  = tensions[m * 8 + d] / masses[m];
    float h0 = home[d];
    float c  = cs[m];
    const float* fp = forces + (size_t)chain * NS + chunk * CHUNK;
    const float* hp = hm     + (size_t)chain * NS + chunk * CHUNK;
    float pos = 0.f, vel = 0.f;
    for (int i = 0; i < CHUNK; ++i) {
        __builtin_prefetch(fp + i + 32, 0, 1);
        float h   = h0 + hp[i] * c;
        float dir = h - pos;
        float acc = fp[i] + a * dir;
        vel = (vel + acc) * DAMP;
        pos += vel;
    }
    zend[(size_t)idx * 2 + 0] = pos;
    zend[(size_t)idx * 2 + 1] = vel;
}

// ---------------------------------------------------------------------------
// Pass 2: per chain, A^128 via 7 squarings; stitch chunk boundary states.
// ---------------------------------------------------------------------------
__global__ void k_boundary(const float* __restrict__ masses,
                           const float* __restrict__ tensions,
                           const float* __restrict__ zend,
                           float* __restrict__ sstart)
{
    int chain = blockIdx.x * blockDim.x + threadIdx.x;   // 2048
    int d = chain & 7, m = (chain >> 3) & 63;
    float a = tensions[m * 8 + d] / masses[m];
    // state s=[pos,vel]; s' = A s + b_t
    float m00 = 1.f - a * DAMP, m01 = DAMP;
    float m10 =      -a * DAMP, m11 = DAMP;
    for (int s = 0; s < 7; ++s) {   // A^(2^7) = A^128
        float t00 = m00 * m00 + m01 * m10;
        float t01 = m00 * m01 + m01 * m11;
        float t10 = m10 * m00 + m11 * m10;
        float t11 = m10 * m01 + m11 * m11;
        m00 = t00; m01 = t01; m10 = t10; m11 = t11;
    }
    float p = 0.f, v = 0.f;
    for (int k = 0; k < NCHUNK; ++k) {
        size_t o = ((size_t)chain * NCHUNK + k) * 2;
        sstart[o] = p; sstart[o + 1] = v;
        float ze0 = zend[o], ze1 = zend[o + 1];
        float np = m00 * p + m01 * v + ze0;
        float nv = m10 * p + m11 * v + ze1;
        p = np; v = nv;
    }
}

// ---------------------------------------------------------------------------
// JAX threefry2x32 (key = PRNGKey(123) = (0,123)), uniform in [-0.01, 0.01)
// ---------------------------------------------------------------------------
__device__ __forceinline__ uint32_t rotl32(uint32_t x, uint32_t r) {
    return (x << r) | (x >> (32u - r));
}

__device__ __forceinline__ float jax_uniform_123(uint32_t gi)
{
    const uint32_t half = 1048576u;               // n/2, n = 4*64*8192
    uint32_t x0, x1;
    if (gi < half) { x0 = gi; x1 = gi + half; }
    else           { x0 = gi - half; x1 = gi; }
    const uint32_t ks0 = 0u, ks1 = 123u, ks2 = 0u ^ 123u ^ 0x1BD11BDAu;
    x0 += ks0; x1 += ks1;
#define TF_R4(ra, rb, rc, rd)                                    \
    x0 += x1; x1 = rotl32(x1, ra); x1 ^= x0;                     \
    x0 += x1; x1 = rotl32(x1, rb); x1 ^= x0;                     \
    x0 += x1; x1 = rotl32(x1, rc); x1 ^= x0;                     \
    x0 += x1; x1 = rotl32(x1, rd); x1 ^= x0;
    TF_R4(13u, 15u, 26u, 6u)  x0 += ks1; x1 += ks2 + 1u;
    TF_R4(17u, 29u, 16u, 24u) x0 += ks2; x1 += ks0 + 2u;
    TF_R4(13u, 15u, 26u, 6u)  x0 += ks0; x1 += ks1 + 3u;
    TF_R4(17u, 29u, 16u, 24u) x0 += ks1; x1 += ks2 + 4u;
    TF_R4(13u, 15u, 26u, 6u)  x0 += ks2; x1 += ks0 + 5u;
#undef TF_R4
    uint32_t bits = (gi < half) ? x0 : x1;
    float u = __uint_as_float((bits >> 9) | 0x3f800000u) - 1.0f;
    return u * 0.02f - 0.01f;
}

// ---------------------------------------------------------------------------
// Pass 3: re-run chunks from exact start state; write displacement, reduce
// over d for recording, compute x = |rec| * noise.
// block = 256 = 8(d) x 32(chunks); grid.x = 512 (bm * 2 halves)
// ---------------------------------------------------------------------------
__global__ void k_finalize(const float* __restrict__ forces,
                           const float* __restrict__ hm,
                           const float* __restrict__ home,
                           const float* __restrict__ cs,
                           const float* __restrict__ masses,
                           const float* __restrict__ tensions,
                           const float* __restrict__ mics,
                           const float* __restrict__ sstart,
                           float* __restrict__ rec_out,
                           float* __restrict__ disp_out,
                           float* __restrict__ x_out)
{
    int tid   = threadIdx.x;
    int bm    = blockIdx.x >> 1;                   // 0..255
    int chunk = (blockIdx.x & 1) * 32 + (tid >> 3);
    int d     = tid & 7;
    int m     = bm & 63;
    int chain = bm * 8 + d;
    float ms   = masses[m];
    float a    = tensions[m * 8 + d] / ms;
    float micv = mics[m * 8 + d];
    float h0   = home[d];
    float c    = cs[m];
    size_t so  = ((size_t)chain * NCHUNK + chunk) * 2;
    float pos = sstart[so], vel = sstart[so + 1];
    const float* fp = forces + (size_t)chain * NS + chunk * CHUNK;
    const float* hp = hm     + (size_t)chain * NS + chunk * CHUNK;
    float* dp = disp_out + (size_t)chain * NS + chunk * CHUNK;
    int tbase = chunk * CHUNK;
    for (int i = 0; i < CHUNK; ++i) {
        float h   = h0 + hp[i] * c;
        float dir = h - pos;
        float acc = fp[i] + a * dir;
        dp[i] = dir;
        float r = ms * acc * micv;                 // mic-weighted ms*acc
        r += __shfl_xor(r, 1, 32);                 // reduce across d (8 lanes)
        r += __shfl_xor(r, 2, 32);
        r += __shfl_xor(r, 4, 32);
        if (d == 0) {
            uint32_t gi = (uint32_t)(bm * NS + tbase + i);
            float noise = jax_uniform_123(gi);
            rec_out[gi] = r;                        // lf == recording (resample is identity)
            x_out[gi]   = fabsf(r) * noise;
        }
        vel = (vel + acc) * DAMP;
        pos += vel;
    }
}

// ---------------------------------------------------------------------------
// FIR conv via WMMA f32_16x16x32_f16.
// One block per (b,m). D[F,j] = sum_k filters[b,F,k] * x[t0+j-k], K=256 in
// 8 chunks of 32.  hf[t] = gain * sum_F mix[m,F]*D[F,t] + rec[t].
// ---------------------------------------------------------------------------
__global__ void k_conv_wmma(const float* __restrict__ filters,   // (B,8,256)
                            const float* __restrict__ bias,      // (1,M,8,1)
                            const float* __restrict__ tfm,       // (M,8,8)
                            const float* __restrict__ hf_gain,   // (1,)
                            const float* __restrict__ x,         // (B*M, NS)
                            const float* __restrict__ rec,       // (B*M, NS)
                            float* __restrict__ hf)              // (B*M, NS)
{
    __shared__ _Float16 xbuf[384];   // window x[t0b-256 .. t0b+127]
    __shared__ float    mixs[16];

    int tid  = threadIdx.x;
    int bm   = blockIdx.x;
    int b    = bm >> 6, m = bm & 63;
    int lane = tid & 31, w = tid >> 5;     // 8 waves, each owns 16 outputs/iter
    int lo   = (lane < 16) ? 1 : 0;
    int N    = lane & 15;

    if (tid < 16) {
        float s = 0.f;
        if (tid < 8) {
            for (int dd = 0; dd < 8; ++dd)
                s += bias[m * 8 + dd] * tfm[(m * 8 + dd) * 8 + tid];
        }
        mixs[tid] = s;                     // rows 8..15 are zero padding
    }

    // A fragments: this code's k order (not AMD's table order) — lanes 0-15 row M=lane hold K 0..15;
    // lanes 16-31 hold K 16..31 (contiguous, matching B).
    v16h afr[8];
    int row = N;                           // F row (8..15 padded with zeros)
#pragma unroll
    for (int p = 0; p < 8; ++p) {
#pragma unroll
        for (int h = 0; h < 16; ++h) {
            int K = h + (lo ? 0 : 16);
            float v = (row < 8) ? filters[((size_t)b * 8 + row) * 256 + p * 32 + K]
                                : 0.f;
            afr[p][h] = (_Float16)v;
        }
    }

    float gain = hf_gain[0];
    const float* xr = x   + (size_t)bm * NS;
    const float* rr = rec + (size_t)bm * NS;
    float*       ho = hf  + (size_t)bm * NS;

    for (int iter = 0; iter < 64; ++iter) {
        int t0b = iter * 128;
        __syncthreads();                   // protect previous window reads
        for (int i = tid; i < 384; i += 256) {
            int gt = t0b - 256 + i;
            xbuf[i] = (_Float16)((gt >= 0) ? xr[gt] : 0.f);   // causal zero-pad
        }
        __syncthreads();

        v8f c = {};
#pragma unroll
        for (int p = 0; p < 8; ++p) {
            // B 32x16 layout: lane holds col N, K = (lo?0:16)+h  =>
            // contiguous reversed run in the LDS window.
            v16h bfr;
            int base = 256 + 16 * w + N - (p * 32 + (lo ? 0 : 16));
#pragma unroll
            for (int h = 0; h < 16; ++h)
                bfr[h] = xbuf[base - h];
            c = __builtin_amdgcn_wmma_f32_16x16x32_f16(
                    false, afr[p], false, bfr, (short)0, c, false, false);
        }

        if (lo) {  // lanes 0-15 hold rows M=0..7 (the valid F rows) in c[0..7]
            float s = 0.f;
#pragma unroll
            for (int r2 = 0; r2 < 8; ++r2) s += mixs[r2] * c[r2];
            int t = t0b + 16 * w + N;
            ho[t] = gain * s + rr[t];
        }
    }
}

// ---------------------------------------------------------------------------
extern "C" void kernel_launch(void* const* d_in, const int* in_sizes, int n_in,
                              void* d_out, int out_size, void* d_ws, size_t ws_size,
                              hipStream_t stream)
{
    (void)in_sizes; (void)n_in; (void)out_size; (void)ws_size;
    const float* forces  = (const float*)d_in[0];
    const float* hm      = (const float*)d_in[1];
    const float* filters = (const float*)d_in[2];
    const float* home    = (const float*)d_in[3];
    const float* cs      = (const float*)d_in[4];
    const float* masses  = (const float*)d_in[5];
    const float* tens    = (const float*)d_in[6];
    const float* mics    = (const float*)d_in[7];
    const float* tfm     = (const float*)d_in[8];
    const float* bias    = (const float*)d_in[9];
    const float* gain    = (const float*)d_in[10];

    float* out  = (float*)d_out;
    float* recO = out + OUT_REC;
    float* disp = out + OUT_DISP;
    float* hfO  = out + OUT_HF;

    float* ws     = (float*)d_ws;
    float* xbuf   = ws + WS_X;
    float* zend   = ws + WS_ZEND;
    float* sstart = ws + WS_SSTART;

    k_chunk_zero<<<512, 256, 0, stream>>>(forces, hm, home, cs, masses, tens, zend);
    k_boundary <<<8,   256, 0, stream>>>(masses, tens, zend, sstart);
    k_finalize <<<512, 256, 0, stream>>>(forces, hm, home, cs, masses, tens, mics,
                                         sstart, recO, disp, xbuf);
    k_conv_wmma<<<256, 256, 0, stream>>>(filters, bias, tfm, gain, xbuf, recO, hfO);
}
